// EncoderLayer_14482629722465
// MI455X (gfx1250) — matrix operand order repaired and checked by reading, not compiled, not run on hardware
//
#include <hip/hip_runtime.h>

typedef __attribute__((ext_vector_type(16))) __bf16 v16bf;
typedef __attribute__((ext_vector_type(2)))  __bf16 v2bf;
typedef __attribute__((ext_vector_type(8)))  float  v8f;
typedef __attribute__((ext_vector_type(4)))  unsigned int v4u;
typedef __attribute__((ext_vector_type(8)))  int v8i;
typedef __attribute__((ext_vector_type(4)))  int v4i;

#define B_  8
#define S_  512
#define H_  1024
#define NH_ 16
#define DK_ 64
#define FF_ 4096
#define BS_ (B_*S_)

static __device__ __forceinline__ v8f wmma_bf16(v16bf a, v16bf b, v8f c) {
  return __builtin_amdgcn_wmma_f32_16x16x32_bf16(false, a, false, b, (short)0, c, false, false);
}

// low 32 bits of flat LDS address = LDS byte offset
static __device__ __forceinline__ unsigned lds_off(const void* p) {
  return (unsigned)(uintptr_t)p;
}

// CDNA5 async global->LDS copy (16B per lane), tracked by ASYNCcnt.
static __device__ __forceinline__ void async_b128(unsigned lds_addr, const void* g) {
  unsigned long long ga = (unsigned long long)(uintptr_t)g;
  asm volatile("global_load_async_to_lds_b128 %0, %1, off" :: "v"(lds_addr), "v"(ga) : "memory");
}

// CDNA5 Tensor Data Mover: DMA a 2-D tile (bf16, row-major, compacted into LDS).
// D# built per ISA 08_async_tensor: group0 = count/lds/global/type, group1 = dims.
// This toolchain exposes the 6-arg builtin (g0, g1, g2, g3, g4, cpol).
static __device__ __forceinline__ void tdm_load_2d(
    unsigned lds_addr, const void* gaddr,
    unsigned tensor_d0, unsigned tensor_d1, unsigned stride_d0,
    unsigned tile_d0, unsigned tile_d1) {
  unsigned long long ga = (unsigned long long)(uintptr_t)gaddr;
  v4u g0;
  g0[0] = 1u;                                             // count=1, user descriptor
  g0[1] = lds_addr;                                       // lds_addr[31:0]
  g0[2] = (unsigned)(ga & 0xffffffffu);                   // global_addr[31:0]
  g0[3] = (unsigned)((ga >> 32) & 0x01ffffffu) | (2u << 30);  // addr[56:32] | type=2
  v8i g1;
  g1[0] = (int)(1u << 16);                                // data_size=1 (2 bytes)
  g1[1] = (int)((tensor_d0 & 0xffffu) << 16);             // tensor_dim0[15:0] @ [63:48]
  g1[2] = (int)((tensor_d0 >> 16) | ((tensor_d1 & 0xffffu) << 16));
  g1[3] = (int)((tensor_d1 >> 16) | (tile_d0 << 16));     // tile_dim0 @ [127:112]
  g1[4] = (int)(tile_d1 & 0xffffu);                       // tile_dim1; tile_dim2=0
  g1[5] = (int)stride_d0;                                 // tensor_dim0_stride[31:0]
  g1[6] = 0;                                              // stride hi / dim1_stride lo
  g1[7] = 0;
  v4i z4 = {0, 0, 0, 0};
  v8i z8 = {0, 0, 0, 0, 0, 0, 0, 0};
  __builtin_amdgcn_tensor_load_to_lds(g0, g1, z4, z4, z8, 0);
}

// ---------------------------------------------------------------------------
// Tiled f32 -> bf16 transpose for weights: in [K][N] f32 -> out [N][K] bf16
// ---------------------------------------------------------------------------
__global__ __launch_bounds__(256) void cvt_transpose_bf16_kernel(const float* __restrict__ in,
                                                                 __bf16* __restrict__ out,
                                                                 int K, int N) {
  __shared__ float tile[64][65];
  int k0 = blockIdx.y * 64, n0 = blockIdx.x * 64;
  int tx = threadIdx.x & 63, ty = threadIdx.x >> 6;
#pragma unroll
  for (int i = ty; i < 64; i += 4)
    tile[i][tx] = in[(size_t)(k0 + i) * N + n0 + tx];
  __syncthreads();
#pragma unroll
  for (int i = ty; i < 64; i += 4)
    out[(size_t)(n0 + i) * K + k0 + tx] = (__bf16)tile[tx][i];
}

// ---------------------------------------------------------------------------
// LayerNorm over H=1024, one row per block (256 threads), bf16 out
// ---------------------------------------------------------------------------
__global__ __launch_bounds__(256) void layernorm_bf16_kernel(const float* __restrict__ x,
                                                             const float* __restrict__ g,
                                                             const float* __restrict__ bta,
                                                             __bf16* __restrict__ y) {
  int row = blockIdx.x;
  int tid = threadIdx.x, lane = tid & 31, wid = tid >> 5;
  const float* xr = x + (size_t)row * H_;
  float vals[4];
  float s = 0.f, s2 = 0.f;
#pragma unroll
  for (int i = 0; i < 4; ++i) {
    float v = xr[tid + 256 * i];
    vals[i] = v; s += v; s2 += v * v;
  }
#pragma unroll
  for (int m = 1; m < 32; m <<= 1) { s += __shfl_xor(s, m, 32); s2 += __shfl_xor(s2, m, 32); }
  __shared__ float sh1[8], sh2[8];
  if (lane == 0) { sh1[wid] = s; sh2[wid] = s2; }
  __syncthreads();
  s = 0.f; s2 = 0.f;
#pragma unroll
  for (int i = 0; i < 8; ++i) { s += sh1[i]; s2 += sh2[i]; }
  float mean = s * (1.f / H_);
  float var = s2 * (1.f / H_) - mean * mean;
  float r = rsqrtf(var + 1e-5f);
#pragma unroll
  for (int i = 0; i < 4; ++i) {
    int c = tid + 256 * i;
    y[(size_t)row * H_ + c] = (__bf16)((vals[i] - mean) * r * g[c] + bta[c]);
  }
}

// ---------------------------------------------------------------------------
// Row softmax over 512 cols, f32 in -> bf16 out
// ---------------------------------------------------------------------------
__global__ __launch_bounds__(256) void softmax512_bf16_kernel(const float* __restrict__ in,
                                                              __bf16* __restrict__ out) {
  int row = blockIdx.x;
  int tid = threadIdx.x, lane = tid & 31, wid = tid >> 5;
  const float* xr = in + (size_t)row * S_;
  float a = xr[tid], b = xr[tid + 256];
  float mx = fmaxf(a, b);
#pragma unroll
  for (int m = 1; m < 32; m <<= 1) mx = fmaxf(mx, __shfl_xor(mx, m, 32));
  __shared__ float shm[8], shs[8];
  if (lane == 0) shm[wid] = mx;
  __syncthreads();
  float gm = shm[0];
#pragma unroll
  for (int i = 1; i < 8; ++i) gm = fmaxf(gm, shm[i]);
  float ea = __expf(a - gm), eb = __expf(b - gm);
  float s = ea + eb;
#pragma unroll
  for (int m = 1; m < 32; m <<= 1) s += __shfl_xor(s, m, 32);
  if (lane == 0) shs[wid] = s;
  __syncthreads();
  float gs = 0.f;
#pragma unroll
  for (int i = 0; i < 8; ++i) gs += shs[i];
  float inv = 1.f / gs;
  out[(size_t)row * S_ + tid] = (__bf16)(ea * inv);
  out[(size_t)row * S_ + tid + 256] = (__bf16)(eb * inv);
}

// ---------------------------------------------------------------------------
// Predicate gather-mean over path-attention output: path[b][d]
// ---------------------------------------------------------------------------
__global__ __launch_bounds__(256) void path_gather_kernel(const float* __restrict__ xp,
                                                          const int* __restrict__ pp,
                                                          float* __restrict__ path) {
  int d = blockIdx.x * 256 + threadIdx.x;
  int b = blockIdx.y;
  const int* ppb = pp + b * S_;
  const float* base = xp + ((size_t)(b * S_)) * H_ + d;
  float acc = 0.f, wsum = 0.f;
  for (int s = 0; s < S_; ++s) {
    if (ppb[s] == 1) {
      float a;
      if (s == 0) {
        a = 0.5f * (base[0] + base[(size_t)H_]);
      } else if (s == 1) {
        a = (base[0] + base[(size_t)H_] + base[(size_t)2 * H_]) * (1.f / 3.f);
      } else if (ppb[s - 2] == 1) {
        int sn = (s + 1 < S_) ? s + 1 : S_ - 1;
        a = 0.5f * (base[(size_t)s * H_] + base[(size_t)sn * H_]);
      } else {
        int sn = (s + 1 < S_) ? s + 1 : S_ - 1;
        a = (base[(size_t)(s - 1) * H_] + base[(size_t)s * H_] + base[(size_t)sn * H_]) * (1.f / 3.f);
      }
      acc += a; wsum += 1.f;
    }
  }
  path[(size_t)b * H_ + d] = acc / wsum;
}

// ---------------------------------------------------------------------------
// Generic bf16 WMMA GEMM, 128x128 tile, K-step 64, double-buffered LDS.
// Tile staging via Tensor Data Mover (one wave issues whole-tile DMA).
// WTRANS=true : W element (k,n) at W[n*ldw+k]  (TDM-staged, contiguous)
// WTRANS=false: W row-major [K][N], ldw=N      (manual transpose-scatter staging)
// MODE 0: bf16 out = (acc + bias[n]) * scale
// MODE 1: f32  out = acc + aux0[z,m,n] + aux1[z,m,n]              (patt)
// MODE 2: f32  out = acc                                           (xp)
// MODE 3: f32  out = aux0[m,n] + 0.5*(acc + bias[n] + aux1[m/S,n]) (combine)
// MODE 4: bf16 out = gelu(acc + bias[n])                           (FFN1)
// MODE 5: f32  out = aux0[m,n] + acc + bias[n]                     (FFN2)
// ---------------------------------------------------------------------------
template <int MODE, bool WTRANS>
__global__ __launch_bounds__(256) void gemm_bf16_wmma_kernel(
    const __bf16* __restrict__ A, size_t strideAz, int lda,
    const __bf16* __restrict__ W, size_t strideWz, int ldw,
    const float* __restrict__ bias,
    const float* __restrict__ aux0, size_t strideAuxz,
    const float* __restrict__ aux1,
    void* __restrict__ Cout, size_t strideCz,
    int M, int N, int K, float scale) {
  __shared__ __align__(16) __bf16 As[2][128][64];  // 2 x 16KB
  __shared__ __align__(16) __bf16 Ws[2][128][64];  // 2 x 16KB, stored [n][k]

  int tid = threadIdx.x, lane = tid & 31, wid = tid >> 5;
  int half = lane >> 4, ln = lane & 15;
  int z = blockIdx.z;
  int m0 = blockIdx.y * 128, n0 = blockIdx.x * 128;
  const __bf16* Ab = A + (size_t)z * strideAz;
  const __bf16* Wb = W + (size_t)z * strideWz;
  int wm = (wid >> 1) * 32, wn = (wid & 1) * 64;

  v8f acc[2][4];
#pragma unroll
  for (int mi = 0; mi < 2; ++mi)
#pragma unroll
    for (int nt = 0; nt < 4; ++nt)
#pragma unroll
      for (int e = 0; e < 8; ++e) acc[mi][nt][e] = 0.f;

  auto stage = [&](int buf, int k0) {
    if (wid == 0) {  // one wave drives the TDM for the A tile (128 rows x 64 k)
      tdm_load_2d(lds_off(&As[buf][0][0]), Ab + (size_t)m0 * lda + k0,
                  (unsigned)lda, (unsigned)M, (unsigned)lda, 64u, 128u);
      if (WTRANS)    // W tile (128 n-rows x 64 k), contiguous in k
        tdm_load_2d(lds_off(&Ws[buf][0][0]), Wb + (size_t)n0 * ldw + k0,
                    (unsigned)ldw, (unsigned)N, (unsigned)ldw, 64u, 128u);
    }
    if (!WTRANS) {  // row-major [K][N]: wide global loads + LDS transpose scatter
      int krow = tid >> 2, ncol = (tid & 3) * 32;
      const uint4* src = (const uint4*)(Wb + (size_t)(k0 + krow) * ldw + n0 + ncol);
      union { uint4 u[4]; __bf16 h[32]; } t;
#pragma unroll
      for (int l = 0; l < 4; ++l) t.u[l] = src[l];
#pragma unroll
      for (int e = 0; e < 32; ++e) Ws[buf][ncol + e][krow] = t.h[e];
    }
  };

  auto load_a_frag = [&](int buf, int rowBase, int kOff) {
    v16bf f;
    int m = rowBase + ln;
    int kb = half ? 8 : 0;
#pragma unroll
    for (int j = 0; j < 8; ++j) {
      int kk = kb + ((j >> 2) << 4) + ((j & 3) << 1);
      v2bf pr = *(const v2bf*)&As[buf][m][kOff + kk];
      f[2 * j] = pr[0]; f[2 * j + 1] = pr[1];
    }
    return f;
  };
  auto load_b_frag = [&](int buf, int nBase, int kOff) {
    v16bf f;
    int n = nBase + ln;
    int kb = half ? 8 : 0;
#pragma unroll
    for (int j = 0; j < 8; ++j) {
      v2bf pr = *(const v2bf*)&Ws[buf][n][kOff + kb + ((j >> 2) << 4) + ((j & 3) << 1)];
      f[2 * j] = pr[0]; f[2 * j + 1] = pr[1];
    }
    return f;
  };

  stage(0, 0);
  int buf = 0;
  for (int k0 = 0; k0 < K; k0 += 64) {
    bool nxt = (k0 + 64 < K);
    if (nxt) stage(buf ^ 1, k0 + 64);  // overlap next-stage DMA with this-stage WMMAs
    if (wid == 0) {
      if (nxt) __builtin_amdgcn_s_wait_tensorcnt(WTRANS ? 2 : 1);
      else     __builtin_amdgcn_s_wait_tensorcnt(0);
    }
    __syncthreads();
#pragma unroll
    for (int kc = 0; kc < 2; ++kc) {
      v16bf a0 = load_a_frag(buf, wm, kc * 32);
      v16bf a1 = load_a_frag(buf, wm + 16, kc * 32);
#pragma unroll
      for (int nt = 0; nt < 4; ++nt) {
        v16bf bf = load_b_frag(buf, wn + nt * 16, kc * 32);
        acc[0][nt] = wmma_bf16(a0, bf, acc[0][nt]);
        acc[1][nt] = wmma_bf16(a1, bf, acc[1][nt]);
      }
    }
    __syncthreads();
    buf ^= 1;
  }

  // epilogue (C layout: VGPR r -> row r + 8*half, col = lane&15 within 16-tile)
#pragma unroll
  for (int mi = 0; mi < 2; ++mi) {
#pragma unroll
    for (int nt = 0; nt < 4; ++nt) {
#pragma unroll
      for (int r = 0; r < 8; ++r) {
        int mg = m0 + wm + mi * 16 + r + 8 * half;
        int ng = n0 + wn + nt * 16 + ln;
        float a = acc[mi][nt][r];
        size_t cidx = (size_t)z * strideCz + (size_t)mg * N + ng;
        if (MODE == 0) {
          ((__bf16*)Cout)[cidx] = (__bf16)((a + bias[ng]) * scale);
        } else if (MODE == 1) {
          size_t aidx = (size_t)z * strideAuxz + (size_t)mg * N + ng;
          ((float*)Cout)[cidx] = a + aux0[aidx] + aux1[aidx];
        } else if (MODE == 2) {
          ((float*)Cout)[cidx] = a;
        } else if (MODE == 3) {
          ((float*)Cout)[cidx] = aux0[cidx] + 0.5f * (a + bias[ng] + aux1[(size_t)(mg / S_) * N + ng]);
        } else if (MODE == 4) {
          float t = a + bias[ng];
          ((__bf16*)Cout)[cidx] = (__bf16)(0.5f * t * (1.0f + erff(t * 0.70710678118f)));
        } else {
          ((float*)Cout)[cidx] = aux0[cidx] + a + bias[ng];
        }
      }
    }
  }
}

// ---------------------------------------------------------------------------
// Fused flash-style self attention. grid (S/64, NH, B), 128 threads (4 waves).
// Block-shared K/V tiles in LDS (K via async-copy engine, V transposed).
// q pre-scaled by DK^-0.5; scores += attn_bias + attention_mask.
// ---------------------------------------------------------------------------
__global__ __launch_bounds__(128) void flash_attn_kernel(
    const __bf16* __restrict__ qg, const __bf16* __restrict__ kg, const __bf16* __restrict__ vg,
    const float* __restrict__ bias, const float* __restrict__ mask,
    __bf16* __restrict__ attv) {
  int b = blockIdx.z, h = blockIdx.y, tile = blockIdx.x;
  int tid = threadIdx.x, wid = tid >> 5, lane = tid & 31;
  int half = lane >> 4, ln = lane & 15;
  int r0 = tile * 64 + wid * 16;

  __shared__ __align__(16) __bf16 Kt[32][64];      // [key][dim]
  __shared__ __align__(16) __bf16 Vt[64][32];      // [dim][key] (transposed)
  __shared__ __align__(16) __bf16 Pst[4][16][32];  // per-wave P staging

  // Q strip as two A fragments (d 0..31 / 32..63)
  v16bf aq0, aq1;
  {
    size_t base = ((size_t)(b * S_) + (r0 + ln)) * H_ + h * DK_;
    int kb = half ? 8 : 0;
#pragma unroll
    for (int j = 0; j < 8; ++j) {
      int d = kb + ((j >> 2) << 4) + ((j & 3) << 1);
      v2bf p0 = *(const v2bf*)(qg + base + d);
      aq0[2 * j] = p0[0]; aq0[2 * j + 1] = p0[1];
      v2bf p1 = *(const v2bf*)(qg + base + 32 + d);
      aq1[2 * j] = p1[0]; aq1[2 * j + 1] = p1[1];
    }
  }

  float mrow[8], lrow[8];
  v8f od[4];
#pragma unroll
  for (int r = 0; r < 8; ++r) { mrow[r] = -1e30f; lrow[r] = 0.f; }
#pragma unroll
  for (int dt = 0; dt < 4; ++dt)
#pragma unroll
    for (int e = 0; e < 8; ++e) od[dt][e] = 0.f;

  for (int j0 = 0; j0 < S_; j0 += 32) {
    // ---- cooperative K/V tile staging ----
    {
      int key = tid >> 2, dcol = (tid & 3) * 16;
      const __bf16* gk = kg + ((size_t)(b * S_) + j0 + key) * H_ + h * DK_ + dcol;
      async_b128(lds_off(&Kt[key][dcol]), gk);
      async_b128(lds_off(&Kt[key][dcol + 8]), gk + 8);
      const uint4* gv = (const uint4*)(vg + ((size_t)(b * S_) + j0 + key) * H_ + h * DK_ + dcol);
      union { uint4 u[2]; __bf16 hh[16]; } t;
      t.u[0] = gv[0]; t.u[1] = gv[1];
#pragma unroll
      for (int e = 0; e < 16; ++e) Vt[dcol + e][key] = t.hh[e];
    }
    asm volatile("s_wait_asynccnt 0" ::: "memory");
    __syncthreads();

    // ---- S = Q.K^T (two 16-key tiles) ----
    v8f sc[2];
#pragma unroll
    for (int t = 0; t < 2; ++t) {
      v16bf bk0, bk1;
      int key = t * 16 + ln;
      int kb = half ? 8 : 0;
#pragma unroll
      for (int j = 0; j < 8; ++j) {
        int d = kb + ((j >> 2) << 4) + ((j & 3) << 1);
        v2bf p0 = *(const v2bf*)&Kt[key][d];
        bk0[2 * j] = p0[0]; bk0[2 * j + 1] = p0[1];
        v2bf p1 = *(const v2bf*)&Kt[key][32 + d];
        bk1[2 * j] = p1[0]; bk1[2 * j + 1] = p1[1];
      }
      v8f zacc;
#pragma unroll
      for (int e = 0; e < 8; ++e) zacc[e] = 0.f;
      zacc = wmma_bf16(aq0, bk0, zacc);
      zacc = wmma_bf16(aq1, bk1, zacc);
      sc[t] = zacc;
    }

    // ---- online softmax ----
    float p[2][8], alpha[8];
#pragma unroll
    for (int r = 0; r < 8; ++r) {
      int row = r0 + r + 8 * half;
      size_t rb = ((size_t)b * S_ + row) * S_ + j0;
      float s0 = sc[0][r] + bias[rb + ln] + mask[rb + ln];
      float s1 = sc[1][r] + bias[rb + 16 + ln] + mask[rb + 16 + ln];
      float tm = fmaxf(s0, s1);
#pragma unroll
      for (int m = 1; m < 16; m <<= 1) tm = fmaxf(tm, __shfl_xor(tm, m, 16));
      float mnew = fmaxf(mrow[r], tm);
      float al = __expf(mrow[r] - mnew);
      float p0 = __expf(s0 - mnew), p1 = __expf(s1 - mnew);
      float rs = p0 + p1;
#pragma unroll
      for (int m = 1; m < 16; m <<= 1) rs += __shfl_xor(rs, m, 16);
      lrow[r] = lrow[r] * al + rs;
      mrow[r] = mnew;
      alpha[r] = al;
      p[0][r] = p0; p[1][r] = p1;
    }
#pragma unroll
    for (int dt = 0; dt < 4; ++dt)
#pragma unroll
      for (int r = 0; r < 8; ++r) od[dt][r] *= alpha[r];

    // ---- stage P (C-frag -> A-frag relayout through LDS) ----
#pragma unroll
    for (int t = 0; t < 2; ++t)
#pragma unroll
      for (int r = 0; r < 8; ++r)
        Pst[wid][r + 8 * half][t * 16 + ln] = (__bf16)p[t][r];
    __syncthreads();
    v16bf ap;
    {
      int kb = half ? 8 : 0;
#pragma unroll
      for (int j = 0; j < 8; ++j) {
        int kk = kb + ((j >> 2) << 4) + ((j & 3) << 1);
        v2bf pr = *(const v2bf*)&Pst[wid][ln][kk];
        ap[2 * j] = pr[0]; ap[2 * j + 1] = pr[1];
      }
    }

    // ---- O += P.V (V fragments from transposed LDS tile) ----
#pragma unroll
    for (int dt = 0; dt < 4; ++dt) {
      v16bf bv;
      int n = dt * 16 + ln;
      int kb = half ? 8 : 0;
#pragma unroll
      for (int j = 0; j < 8; ++j) {
        v2bf pr = *(const v2bf*)&Vt[n][kb + ((j >> 2) << 4) + ((j & 3) << 1)];
        bv[2 * j] = pr[0]; bv[2 * j + 1] = pr[1];
      }
      od[dt] = wmma_bf16(ap, bv, od[dt]);
    }
    __syncthreads();  // protect Kt/Vt/Pst before next stage
  }

#pragma unroll
  for (int dt = 0; dt < 4; ++dt)
#pragma unroll
    for (int r = 0; r < 8; ++r) {
      int row = r0 + r + 8 * half;
      attv[((size_t)(b * S_) + row) * H_ + h * DK_ + dt * 16 + ln] = (__bf16)(od[dt][r] / lrow[r]);
    }
}

// ---------------------------------------------------------------------------
// Host orchestration
// ---------------------------------------------------------------------------
extern "C" void kernel_launch(void* const* d_in, const int* in_sizes, int n_in,
                              void* d_out, int out_size, void* d_ws, size_t ws_size,
                              hipStream_t stream) {
  (void)in_sizes; (void)n_in; (void)out_size; (void)ws_size;
  const float* x    = (const float*)d_in[0];
  const int*   pp   = (const int*)d_in[1];
  // d_in[2] variable_tags: unused
  const float* attn_bias = (const float*)d_in[3];
  const float* attn_mask = (const float*)d_in[4];
  const float* atomg = (const float*)d_in[5];
  const float* varg  = (const float*)d_in[6];
  const float* ln1g = (const float*)d_in[7];
  const float* ln1b = (const float*)d_in[8];
  const float* wq = (const float*)d_in[9];   const float* bq = (const float*)d_in[10];
  const float* wk = (const float*)d_in[11];  const float* bk = (const float*)d_in[12];
  const float* wv = (const float*)d_in[13];  const float* bv = (const float*)d_in[14];
  const float* wo = (const float*)d_in[15];  const float* bo = (const float*)d_in[16];
  const float* pqw = (const float*)d_in[17]; const float* pqb = (const float*)d_in[18];
  const float* pkw = (const float*)d_in[19]; const float* pkb = (const float*)d_in[20];
  const float* pvw = (const float*)d_in[21]; const float* pvb = (const float*)d_in[22];
  const float* ln2g = (const float*)d_in[23];
  const float* ln2b = (const float*)d_in[24];
  const float* w1 = (const float*)d_in[25];  const float* b1 = (const float*)d_in[26];
  const float* w2 = (const float*)d_in[27];  const float* b2 = (const float*)d_in[28];

  const size_t H2 = (size_t)H_ * H_;
  const size_t HF = (size_t)H_ * FF_;
  const size_t BSH = (size_t)BS_ * H_;
  const size_t BSS = (size_t)B_ * S_ * S_;
  const size_t BSF = (size_t)BS_ * FF_;

  char* p = (char*)d_ws;
  auto alloc = [&](size_t bytes) -> char* {
    char* r = p;
    p += (bytes + 255) & ~(size_t)255;
    return r;
  };

  // transposed bf16 weights: [N][K]
  __bf16* wqT  = (__bf16*)alloc(H2 * 2);
  __bf16* wkT  = (__bf16*)alloc(H2 * 2);
  __bf16* wvT  = (__bf16*)alloc(H2 * 2);
  __bf16* woT  = (__bf16*)alloc(H2 * 2);
  __bf16* pqwT = (__bf16*)alloc(H2 * 2);
  __bf16* pkwT = (__bf16*)alloc(H2 * 2);
  __bf16* pvwT = (__bf16*)alloc(H2 * 2);
  __bf16* w1T  = (__bf16*)alloc(HF * 2);  // [FF][H]
  __bf16* w2T  = (__bf16*)alloc(HF * 2);  // [H][FF]
  __bf16* y_bf   = (__bf16*)alloc(BSH * 2);
  __bf16* q_bf   = (__bf16*)alloc(BSH * 2);
  __bf16* k_bf   = (__bf16*)alloc(BSH * 2);
  __bf16* v_bf   = (__bf16*)alloc(BSH * 2);
  __bf16* pq_bf  = (__bf16*)alloc(BSH * 2);
  __bf16* pk_bf  = (__bf16*)alloc(BSH * 2);
  __bf16* pv_bf  = (__bf16*)alloc(BSH * 2);
  __bf16* attv_bf = (__bf16*)alloc(BSH * 2);
  __bf16* y2_bf  = (__bf16*)alloc(BSH * 2);
  __bf16* hmid_bf = (__bf16*)alloc(BSF * 2);
  __bf16* patt_bf = (__bf16*)alloc(BSS * 2);
  float*  patt_f  = (float*)alloc(BSS * 4);
  float*  xp_f    = (float*)alloc(BSH * 4);
  float*  x1_f    = (float*)alloc(BSH * 4);
  float*  path_f  = (float*)alloc((size_t)B_ * H_ * 4);

  auto cvtT = [&](const float* src, __bf16* dst, int K, int N) {
    cvt_transpose_bf16_kernel<<<dim3(N / 64, K / 64), 256, 0, stream>>>(src, dst, K, N);
  };
  cvtT(wq, wqT, H_, H_);   cvtT(wk, wkT, H_, H_);   cvtT(wv, wvT, H_, H_);
  cvtT(wo, woT, H_, H_);   cvtT(pqw, pqwT, H_, H_); cvtT(pkw, pkwT, H_, H_);
  cvtT(pvw, pvwT, H_, H_); cvtT(w1, w1T, H_, FF_);  cvtT(w2, w2T, FF_, H_);

  // LN1
  layernorm_bf16_kernel<<<BS_, 256, 0, stream>>>(x, ln1g, ln1b, y_bf);

  const float qscale = 0.125f;    // DK^-0.5
  const float pscale = 0.03125f;  // H^-0.5
  dim3 gProj(H_ / 128, BS_ / 128, 1);

  gemm_bf16_wmma_kernel<0, true><<<gProj, 256, 0, stream>>>(
      y_bf, 0, H_, wqT, 0, H_, bq, nullptr, 0, nullptr, q_bf, 0, BS_, H_, H_, qscale);
  gemm_bf16_wmma_kernel<0, true><<<gProj, 256, 0, stream>>>(
      y_bf, 0, H_, wkT, 0, H_, bk, nullptr, 0, nullptr, k_bf, 0, BS_, H_, H_, 1.0f);
  gemm_bf16_wmma_kernel<0, true><<<gProj, 256, 0, stream>>>(
      y_bf, 0, H_, wvT, 0, H_, bv, nullptr, 0, nullptr, v_bf, 0, BS_, H_, H_, 1.0f);
  gemm_bf16_wmma_kernel<0, true><<<gProj, 256, 0, stream>>>(
      y_bf, 0, H_, pqwT, 0, H_, pqb, nullptr, 0, nullptr, pq_bf, 0, BS_, H_, H_, pscale);
  gemm_bf16_wmma_kernel<0, true><<<gProj, 256, 0, stream>>>(
      y_bf, 0, H_, pkwT, 0, H_, pkb, nullptr, 0, nullptr, pk_bf, 0, BS_, H_, H_, 1.0f);
  gemm_bf16_wmma_kernel<0, true><<<gProj, 256, 0, stream>>>(
      y_bf, 0, H_, pvwT, 0, H_, pvb, nullptr, 0, nullptr, pv_bf, 0, BS_, H_, H_, 1.0f);

  // fused self-attention
  flash_attn_kernel<<<dim3(S_ / 64, NH_, B_), 128, 0, stream>>>(
      q_bf, k_bf, v_bf, attn_bias, attn_mask, attv_bf);

  // path attention: patt = pq.pkT + atom + var (pk is naturally "transposed" W)
  gemm_bf16_wmma_kernel<1, true><<<dim3(S_ / 128, S_ / 128, B_), 256, 0, stream>>>(
      pq_bf, (size_t)S_ * H_, H_, pk_bf, (size_t)S_ * H_, H_, nullptr,
      atomg, (size_t)S_ * S_, varg, patt_f, (size_t)S_ * S_, S_, S_, H_, 1.0f);
  softmax512_bf16_kernel<<<BS_, 256, 0, stream>>>(patt_f, patt_bf);
  // xp = patt . pv
  gemm_bf16_wmma_kernel<2, false><<<dim3(H_ / 128, S_ / 128, B_), 256, 0, stream>>>(
      patt_bf, (size_t)S_ * S_, S_, pv_bf, (size_t)S_ * H_, H_, nullptr,
      nullptr, 0, nullptr, xp_f, (size_t)S_ * H_, S_, H_, S_, 1.0f);
  path_gather_kernel<<<dim3(H_ / 256, B_), 256, 0, stream>>>(xp_f, pp, path_f);

  // x1 = x + 0.5*(attv.wo + bo + path[b])
  gemm_bf16_wmma_kernel<3, true><<<gProj, 256, 0, stream>>>(
      attv_bf, 0, H_, woT, 0, H_, bo, x, 0, path_f, x1_f, 0, BS_, H_, H_, 1.0f);

  // LN2 + FFN
  layernorm_bf16_kernel<<<BS_, 256, 0, stream>>>(x1_f, ln2g, ln2b, y2_bf);
  gemm_bf16_wmma_kernel<4, true><<<dim3(FF_ / 128, BS_ / 128, 1), 256, 0, stream>>>(
      y2_bf, 0, H_, w1T, 0, H_, b1, nullptr, 0, nullptr, hmid_bf, 0, BS_, FF_, H_, 1.0f);
  gemm_bf16_wmma_kernel<5, true><<<dim3(H_ / 128, BS_ / 128, 1), 256, 0, stream>>>(
      hmid_bf, 0, FF_, w2T, 0, FF_, b2, x1_f, 0, nullptr, (float*)d_out, 0, BS_, H_, FF_, 1.0f);
}
